// SURGE_74388833567110
// MI455X (gfx1250) — matrix operand order repaired and checked by reading, not compiled, not run on hardware
//
#include <hip/hip_runtime.h>

typedef __attribute__((ext_vector_type(16))) _Float16 v16h;
typedef __attribute__((ext_vector_type(8)))  float    v8f;

#define NNODES 200000
#define NEDGES 800000
#define NGRAPH 2000
#define PNODES 100
#define FDIM   24
#define NPAD   200064   // 1563 * 128
#define NCHUNK 782      // ceil(NNODES/256)

struct Ptr6  { const float* p[6];  };
struct Ptr12 { const float* p[12]; };

__device__ __forceinline__ float leakyf(float x) { return x > 0.f ? x : 0.2f * x; }
__device__ __forceinline__ float sigf(float x)   { return 1.f / (1.f + __expf(-x)); }

// ---- WMMA helpers -------------------------------------------------------
// A-frag (16x32 f16): lane L holds row M=L%16; element j -> K = (L<16?0:8)+j (j<8),
// K = (L<16?16:24)+(j-8) (j>=8).
__device__ __forceinline__ v16h load_a_frag(const float* A, int rowbase, int lane) {
  int row = rowbase + (lane & 15);
  const float* p = A + (size_t)row * 32 + ((lane & 16) ? 8 : 0);
  float4 a0 = *(const float4*)(p);
  float4 a1 = *(const float4*)(p + 4);
  float4 a2 = *(const float4*)(p + 16);
  float4 a3 = *(const float4*)(p + 20);
  v16h f;
  f[0]=(_Float16)a0.x; f[1]=(_Float16)a0.y; f[2]=(_Float16)a0.z; f[3]=(_Float16)a0.w;
  f[4]=(_Float16)a1.x; f[5]=(_Float16)a1.y; f[6]=(_Float16)a1.z; f[7]=(_Float16)a1.w;
  f[8]=(_Float16)a2.x; f[9]=(_Float16)a2.y; f[10]=(_Float16)a2.z; f[11]=(_Float16)a2.w;
  f[12]=(_Float16)a3.x; f[13]=(_Float16)a3.y; f[14]=(_Float16)a3.z; f[15]=(_Float16)a3.w;
  return f;
}

__device__ __forceinline__ v16h frag_from_lds(const float (*sm)[32], int rowbase, int lane) {
  int row = rowbase + (lane & 15);
  int off = (lane & 16) ? 8 : 0;
  v16h f;
#pragma unroll
  for (int j = 0; j < 8; ++j) f[j]     = (_Float16)sm[row][off + j];
#pragma unroll
  for (int j = 0; j < 8; ++j) f[8 + j] = (_Float16)sm[row][off + 16 + j];
  return f;
}

__device__ __forceinline__ v8f wmma_f16(v16h a, v16h b, v8f c) {
  return __builtin_amdgcn_wmma_f32_16x16x32_f16(false, a, false, b, (short)0, c,
                                                false, false);
}

// ---- Weight packing -----------------------------------------------------
// B-frag (32x16 f16): lane L holds column N=L%16; element j -> K=(L<16?0:8)+j (j<8), K=(L<16?16:24)+(j-8) (j>=8).
__global__ void prep_msgw(Ptr6 w6, _Float16* out) {
  int tid = blockIdx.x * blockDim.x + threadIdx.x;
  if (tid >= 160 * 2 * 32) return;
  int lane = tid & 31;
  int t    = (tid >> 5) & 1;
  int S    = tid >> 6;                 // global step index 0..159
  int br = S / 80;
  int s2 = S % 80;
  int layer, step;
  if (s2 < 32)      { layer = 0; step = s2; }
  else if (s2 < 64) { layer = 1; step = s2 - 32; }
  else              { layer = 2; step = s2 - 64; }
  int d = (layer == 0) ? 24 : 32;
  const float* Wl = w6.p[br * 3 + layer] + (size_t)step * d * d;
  int ncol = t * 16 + (lane & 15);
  int kb   = (lane < 16) ? 0 : 8;
  _Float16* o = out + ((size_t)S * 2 + t) * 512 + lane * 16;
#pragma unroll
  for (int j = 0; j < 16; ++j) {
    int K = kb + j + (j >= 8 ? 8 : 0);
    float v = (K < d && ncol < d) ? Wl[K * d + ncol] : 0.f;
    o[j] = (_Float16)v;
  }
}

__global__ void prep_gih(Ptr12 p12, _Float16* out) {
  int tid = blockIdx.x * blockDim.x + threadIdx.x;
  if (tid >= 12 * 6 * 32) return;
  int lane = tid & 31;
  int t    = (tid >> 5) % 6;
  int M    = tid / (32 * 6);           // 0..11
  int layer = (M % 6) / 2;
  int d = (layer == 0) ? 24 : 32;
  const float* Wm = p12.p[M];          // (3d, d) row-major
  int jcol = t * 16 + (lane & 15);
  int blk  = jcol >> 5;
  int cc   = jcol & 31;
  int kb   = (lane < 16) ? 0 : 8;
  _Float16* o = out + (size_t)M * 3072 + (size_t)t * 512 + lane * 16;
#pragma unroll
  for (int j = 0; j < 16; ++j) {
    int K = kb + j + (j >= 8 ? 8 : 0);
    float v = (K < d && cc < d) ? Wm[(blk * d + cc) * d + K] : 0.f;
    o[j] = (_Float16)v;
  }
}

// ---- CSR build (once per launch; edge_index is constant) ---------------
__global__ void csr_zero(int* cnt) {
  int n = blockIdx.x * blockDim.x + threadIdx.x;
  if (n < NNODES) cnt[n] = 0;
}
__global__ void csr_count(const int* __restrict__ ei, int* __restrict__ cnt) {
  int e = blockIdx.x * blockDim.x + threadIdx.x;
  if (e < NEDGES) atomicAdd(&cnt[ei[NEDGES + e]], 1);
}
__global__ void csr_chunksum(const int* __restrict__ cnt, int* __restrict__ csum) {
  int c = blockIdx.x * blockDim.x + threadIdx.x;
  if (c >= NCHUNK) return;
  int s = 0;
  for (int k = 0; k < 256; ++k) {
    int n = c * 256 + k;
    if (n < NNODES) s += cnt[n];
  }
  csum[c] = s;
}
__global__ void csr_scan(const int* __restrict__ csum, int* __restrict__ coff,
                         int* __restrict__ rp) {
  int acc = 0;
  for (int c = 0; c < NCHUNK; ++c) { coff[c] = acc; acc += csum[c]; }
  rp[NNODES] = acc;   // == NEDGES
}
__global__ void csr_rowptr(const int* __restrict__ cnt, const int* __restrict__ coff,
                           int* __restrict__ rp) {
  int n = blockIdx.x * blockDim.x + threadIdx.x;
  if (n >= NNODES) return;
  int c = n >> 8;
  int s = coff[c];
  for (int k = c << 8; k < n; ++k) s += cnt[k];
  rp[n] = s;
}
__global__ void csr_cursor(const int* __restrict__ rp, int* __restrict__ cur) {
  int n = blockIdx.x * blockDim.x + threadIdx.x;
  if (n < NNODES) cur[n] = rp[n];
}
__global__ void csr_fill(const int* __restrict__ ei, int* __restrict__ cur,
                         int* __restrict__ csrc) {
  int e = blockIdx.x * blockDim.x + threadIdx.x;
  if (e >= NEDGES) return;
  int pos = atomicAdd(&cur[ei[NEDGES + e]], 1);
  csrc[pos] = ei[e];
}

// ---- Pipeline kernels ---------------------------------------------------
__global__ void init_h(const float* x, float* h) {
  int tid = blockIdx.x * blockDim.x + threadIdx.x;   // NPAD*32 threads
  int n = tid >> 5, c = tid & 31;
  float v = 0.f;
  if (n < NNODES && c < FDIM) v = x[(size_t)n * FDIM + c];
  h[tid] = v;
}

// hW = h @ Wl (NPAD x 32 @ 32x32)
__global__ __launch_bounds__(128) void gemm_msg(const float* __restrict__ h,
                                                const _Float16* __restrict__ Bp,
                                                float* __restrict__ hW) {
  int lane = threadIdx.x & 31, wave = threadIdx.x >> 5;
  int rowbase = blockIdx.x * 128 + wave * 16;
  v16h a = load_a_frag(h, rowbase, lane);
  int lo = lane & 15, hi = (lane & 16) ? 8 : 0;
#pragma unroll
  for (int t = 0; t < 2; ++t) {
    v16h b = *(const v16h*)(Bp + (size_t)t * 512 + lane * 16);
    v8f c = {};
    c = wmma_f16(a, b, c);
    float* o = hW + (size_t)(rowbase + hi) * 32 + t * 16 + lo;
#pragma unroll
    for (int r = 0; r < 8; ++r) o[(size_t)r * 32] = c[r];
  }
}

// Fused: CSR gather (m in LDS) -> GI/GH via 12 WMMAs -> gates -> h in place
__global__ __launch_bounds__(128) void gru_fused(const int* __restrict__ rp,
                                                 const int* __restrict__ csrc,
                                                 const float* __restrict__ hW,
                                                 float* __restrict__ h,
                                                 const _Float16* __restrict__ Bih,
                                                 const _Float16* __restrict__ Bhh,
                                                 const float* __restrict__ bih,
                                                 const float* __restrict__ bhh,
                                                 int d, int apply_leaky) {
  __shared__ float msh[64][32];        // gathered messages for 64 rows (8 KB)
  __shared__ float gbuf[4][16][192];   // per wave: 16 rows x (GI 96 | GH 96)
  int tid = threadIdx.x;
  int lane = tid & 31, wave = tid >> 5;
  int row0 = blockIdx.x * 64;

  // phase 0: atomic-free message gather, m rows -> LDS
#pragma unroll
  for (int q = 0; q < 4; ++q) {
    int idx = tid + q * 128;           // 0..511
    int lr = idx >> 3, cg = (idx & 7) << 2;
    int n = row0 + lr;
    float4 acc = make_float4(0.f, 0.f, 0.f, 0.f);
    if (n < NNODES) {
      int e1 = rp[n + 1];
      for (int e = rp[n]; e < e1; ++e) {
        int s = csrc[e];
        float4 v = *(const float4*)(hW + (size_t)s * 32 + cg);
        acc.x += v.x; acc.y += v.y; acc.z += v.z; acc.w += v.w;
      }
    }
    *(float4*)&msh[lr][cg] = acc;
  }
  __syncthreads();

  // phase 1: GI = m @ Bih, GH = h @ Bhh
  int rowbase = row0 + wave * 16;
  v16h am = frag_from_lds(msh, wave * 16, lane);
  v16h ah = load_a_frag(h, rowbase, lane);
  int lo = lane & 15, hi = (lane & 16) ? 8 : 0;
#pragma unroll
  for (int t = 0; t < 6; ++t) {
    v16h b1 = *(const v16h*)(Bih + (size_t)t * 512 + lane * 16);
    v8f c = {};
    c = wmma_f16(am, b1, c);
#pragma unroll
    for (int r = 0; r < 8; ++r) gbuf[wave][hi + r][t * 16 + lo] = c[r];
    v16h b2 = *(const v16h*)(Bhh + (size_t)t * 512 + lane * 16);
    v8f c2 = {};
    c2 = wmma_f16(ah, b2, c2);
#pragma unroll
    for (int r = 0; r < 8; ++r) gbuf[wave][hi + r][96 + t * 16 + lo] = c2[r];
  }
  __syncthreads();

  // phase 2: GRU gates (channel = lane)
  int c = lane;
  bool act = (c < d);
  float bi_r = 0.f, bi_z = 0.f, bi_n = 0.f, bh_r = 0.f, bh_z = 0.f, bh_n = 0.f;
  if (act) {
    bi_r = bih[c]; bi_z = bih[d + c]; bi_n = bih[2 * d + c];
    bh_r = bhh[c]; bh_z = bhh[d + c]; bh_n = bhh[2 * d + c];
  }
  for (int i = 0; i < 16; ++i) {
    int row = rowbase + i;
    float outv = 0.f;
    if (act) {
      float gr = gbuf[wave][i][c]       + bi_r;
      float gz = gbuf[wave][i][32 + c]  + bi_z;
      float gn = gbuf[wave][i][64 + c]  + bi_n;
      float hr = gbuf[wave][i][96 + c]  + bh_r;
      float hz = gbuf[wave][i][128 + c] + bh_z;
      float hn = gbuf[wave][i][160 + c] + bh_n;
      float hv = h[(size_t)row * 32 + c];
      float r = sigf(gr + hr);
      float z = sigf(gz + hz);
      float ng = tanhf(gn + r * hn);
      outv = (1.f - z) * ng + z * hv;
      if (apply_leaky) outv = leakyf(outv);
    }
    h[(size_t)row * 32 + c] = outv;
  }
}

__global__ void pool_kernel(const float* __restrict__ h, float* __restrict__ pool) {
  int tid = blockIdx.x * blockDim.x + threadIdx.x;  // NGRAPH*32 threads
  int g = tid >> 5, c = tid & 31;
  float s = 0.f;
  const float* p = h + (size_t)g * PNODES * 32 + c;
  for (int k = 0; k < PNODES; ++k) s += p[(size_t)k * 32];
  pool[(size_t)g * 32 + c] = s * (1.f / PNODES);
}

__global__ void head_mlp(const float* __restrict__ X, int rows,
                         const float* __restrict__ W1, const float* __restrict__ b1,
                         const float* __restrict__ W2, const float* __restrict__ b2,
                         int odim, float* __restrict__ out, int nmolmap) {
  int r = blockIdx.x * blockDim.x + threadIdx.x;
  if (r >= rows) return;
  int xrow = nmolmap ? ((r / 90) * PNODES + (r % 90)) : r;
  const float* xp = X + (size_t)xrow * 32;
  float hb[16];
#pragma unroll
  for (int j = 0; j < 16; ++j) {
    float s = b1[j];
#pragma unroll
    for (int k = 0; k < 32; ++k) s += xp[k] * W1[j * 32 + k];
    hb[j] = leakyf(s);
  }
  for (int o = 0; o < odim; ++o) {
    float s = b2[o];
#pragma unroll
    for (int j = 0; j < 16; ++j) s += hb[j] * W2[o * 16 + j];
    out[(size_t)r * odim + o] = leakyf(s);
  }
}

// ---- Host orchestration -------------------------------------------------
extern "C" void kernel_launch(void* const* d_in, const int* in_sizes, int n_in,
                              void* d_out, int out_size, void* d_ws, size_t ws_size,
                              hipStream_t stream) {
  (void)in_sizes; (void)n_in; (void)out_size; (void)ws_size;
  const float* x  = (const float*)d_in[0];
  const int*   ei = (const int*)d_in[1];

  float* ws   = (float*)d_ws;
  float* h    = ws;
  float* hW   = ws + (size_t)NPAD * 32;
  float* pool = ws + (size_t)NPAD * 32 * 2;
  _Float16* msgw = (_Float16*)(pool + (size_t)NGRAPH * 32);  // 160*1024 halves
  _Float16* gihw = msgw + (size_t)160 * 1024;                // 12*3072 halves
  int* ibase = (int*)(gihw + (size_t)12 * 3072);
  int* rp    = ibase;                      // NNODES+1
  int* cur   = rp + (NNODES + 1);          // NNODES (also deg counts)
  int* csum  = cur + NNODES;               // NCHUNK
  int* coff  = csum + NCHUNK;              // NCHUNK
  int* csrc  = coff + NCHUNK;              // NEDGES

  // params: d_in[3..] in insertion order: per branch/layer {W,Wih,Whh,bih,bhh}
  Ptr6 w6; Ptr12 p12;
  for (int br = 0; br < 2; ++br)
    for (int li = 0; li < 3; ++li) {
      int base = 3 + br * 15 + li * 5;
      w6.p[br * 3 + li]          = (const float*)d_in[base + 0];
      p12.p[br * 6 + li * 2]     = (const float*)d_in[base + 1];
      p12.p[br * 6 + li * 2 + 1] = (const float*)d_in[base + 2];
    }
  prep_msgw<<<40, 256, 0, stream>>>(w6, msgw);
  prep_gih<<<9, 256, 0, stream>>>(p12, gihw);

  // CSR build (once per launch)
  csr_zero<<<782, 256, 0, stream>>>(cur);
  csr_count<<<3125, 256, 0, stream>>>(ei, cur);
  csr_chunksum<<<4, 256, 0, stream>>>(cur, csum);
  csr_scan<<<1, 1, 0, stream>>>(csum, coff, rp);
  csr_rowptr<<<782, 256, 0, stream>>>(cur, coff, rp);
  csr_cursor<<<782, 256, 0, stream>>>(rp, cur);
  csr_fill<<<3125, 256, 0, stream>>>(ei, cur, csrc);

  float* out       = (float*)d_out;
  float* out_t     = out;               // [G,2]
  float* out_nmol  = out + 4000;        // [G*90,1]
  float* out_nfull = out + 184000;      // [N,1]
  float* out_b     = out + 384000;      // [G,3]
  float* out_v     = out + 390000;      // [G,1]

  const int LD[3] = {24, 32, 32};
  const int LL[3] = {32, 32, 16};
  int S = 0;
  for (int br = 0; br < 2; ++br) {
    init_h<<<25008, 256, 0, stream>>>(x, h);
    for (int li = 0; li < 3; ++li) {
      int d = LD[li], L = LL[li];
      int base = 3 + br * 15 + li * 5;
      const float* bih = (const float*)d_in[base + 3];
      const float* bhh = (const float*)d_in[base + 4];
      const _Float16* Bih = gihw + (size_t)(br * 6 + li * 2 + 0) * 3072;
      const _Float16* Bhh = gihw + (size_t)(br * 6 + li * 2 + 1) * 3072;
      for (int s = 0; s < L; ++s, ++S) {
        gemm_msg<<<1563, 128, 0, stream>>>(h, msgw + (size_t)S * 1024, hW);
        gru_fused<<<3126, 128, 0, stream>>>(rp, csrc, hW, h, Bih, Bhh, bih, bhh,
                                            d, (s == L - 1) ? 1 : 0);
      }
    }
    pool_kernel<<<250, 256, 0, stream>>>(h, pool);
    if (br == 0) {
      head_mlp<<<8, 256, 0, stream>>>(pool, NGRAPH,
          (const float*)d_in[33], (const float*)d_in[34],
          (const float*)d_in[35], (const float*)d_in[36], 2, out_t, 0);
      head_mlp<<<704, 256, 0, stream>>>(h, NGRAPH * 90,
          (const float*)d_in[37], (const float*)d_in[38],
          (const float*)d_in[39], (const float*)d_in[40], 1, out_nmol, 1);
      head_mlp<<<782, 256, 0, stream>>>(h, NNODES,
          (const float*)d_in[41], (const float*)d_in[42],
          (const float*)d_in[43], (const float*)d_in[44], 1, out_nfull, 0);
      head_mlp<<<8, 256, 0, stream>>>(pool, NGRAPH,
          (const float*)d_in[45], (const float*)d_in[46],
          (const float*)d_in[47], (const float*)d_in[48], 3, out_b, 0);
    } else {
      head_mlp<<<8, 256, 0, stream>>>(pool, NGRAPH,
          (const float*)d_in[49], (const float*)d_in[50],
          (const float*)d_in[51], (const float*)d_in[52], 1, out_v, 0);
    }
  }
}
